// SimiMultiHeadedAttention_12395275616457
// MI455X (gfx1250) — matrix operand order repaired and checked by reading, not compiled, not run on hardware
//
#include <hip/hip_runtime.h>

typedef _Float16 h16;
typedef __attribute__((ext_vector_type(16))) _Float16 v16h;
typedef __attribute__((ext_vector_type(8)))  float    v8f;
typedef __attribute__((ext_vector_type(4)))  unsigned int u32x4;
typedef __attribute__((ext_vector_type(8)))  int       i32x8;
typedef __attribute__((ext_vector_type(4)))  int       i32x4;

#define NHEADS  16
#define DMODEL  1024
#define DKH     64
#define TSEQ    2048
#define BHTOT   32          // B * NHEADS
#define MROWS   4096        // B * TSEQ

#define WMMA_F16(a, b, c) \
  __builtin_amdgcn_wmma_f32_16x16x32_f16(false, (a), false, (b), (short)0, (c), false, false)

// ---------------------------------------------------------------------------
// LDS fragment loaders for V_WMMA_F32_16X16X32_F16 (wave32 layouts, ISA 7.12.2)
// ---------------------------------------------------------------------------
__device__ __forceinline__ v16h frag_a(const h16* base, int stride) {
  int lane = threadIdx.x & 31;
  int row  = lane & 15;
  int k0   = (lane >> 4) << 3;      // 0 or 8
  union { v16h h; uint4 q[2]; } f;
  const h16* p = base + row * stride;
  f.q[0] = *(const uint4*)(p + k0);
  f.q[1] = *(const uint4*)(p + k0 + 16);
  return f.h;
}

__device__ __forceinline__ v16h frag_b(const h16* base, int stride) {
  int lane = threadIdx.x & 31;
  int col  = lane & 15;
  int k0   = (lane >> 4) << 3;      // 0 or 8
  union { v16h h; uint4 q[2]; } f;
  const h16* p = base + col * stride;
  f.q[0] = *(const uint4*)(p + k0);
  f.q[1] = *(const uint4*)(p + k0 + 16);
  return f.h;
}

// ---------------------------------------------------------------------------
// Tensor Data Mover: 2D f16 tile  global -> LDS  (D# per ISA 08_async_tensor §8)
// ISA §10.2: flat LDS aperture takes the LDS address from the low address bits, so the low 32 bits
// of a flat __shared__ pointer are the LDS byte address for D#.lds_addr.
// This toolchain exposes the 6-arg builtin (g0, g1, g2, g3, g4, cpol).
// ---------------------------------------------------------------------------
__device__ __forceinline__ unsigned lds_addr_of(const void* p) {
  return (unsigned)(unsigned long long)p;
}

__device__ __forceinline__ void tdm_load_2d_f16(unsigned ldsAddr, const void* gaddr,
                                                unsigned tensorW, unsigned tensorH,
                                                unsigned strideElems,
                                                unsigned tileW, unsigned tileH) {
  unsigned long long ga = (unsigned long long)gaddr;
  u32x4 g0;
  g0[0] = 1u;                                        // count=1, user descriptor
  g0[1] = ldsAddr;                                   // lds_addr (bytes)
  g0[2] = (unsigned)(ga & 0xFFFFFFFFu);              // global_addr[31:0]
  g0[3] = (unsigned)((ga >> 32) & 0x01FFFFFFu) | (2u << 30);  // addr[56:32], type=2
  i32x8 g1;
  g1[0] = (int)(1u << 16);                           // wg_mask=0, data_size=1 (2B)
  g1[1] = (int)((tensorW & 0xFFFFu) << 16);          // tensor_dim0[15:0] @63:48
  g1[2] = (int)(((tensorW >> 16) & 0xFFFFu) |        // tensor_dim0[31:16]
                ((tensorH & 0xFFFFu) << 16));        // tensor_dim1[15:0]
  g1[3] = (int)(((tensorH >> 16) & 0xFFFFu) |        // tensor_dim1[31:16]
                ((tileW & 0xFFFFu) << 16));          // tile_dim0 @127:112
  g1[4] = (int)(tileH & 0xFFFFu);                    // tile_dim1, tile_dim2=0
  g1[5] = (int)strideElems;                          // tensor_dim0_stride[31:0]
  g1[6] = 0;                                         // stride high bits, dim1_stride lo
  g1[7] = 0;
  i32x4 z4 = {0, 0, 0, 0};                           // groups 2/3 unused (2D)
  i32x8 z8 = {0, 0, 0, 0, 0, 0, 0, 0};
  __builtin_amdgcn_tensor_load_to_lds(g0, g1, z4, z4, z8, 0);
}

// ---------------------------------------------------------------------------
// Weight prep: W f32 [K=1024][N=1024] (in,out)  ->  WT f16 [N][K]
// ---------------------------------------------------------------------------
__global__ __launch_bounds__(256) void wconvT(const float* __restrict__ W,
                                              h16* __restrict__ WT) {
  int idx = blockIdx.x * 256 + threadIdx.x;  // n*1024 + k
  int k = idx & 1023;
  int n = idx >> 10;
  WT[idx] = (h16)W[k * 1024 + n];
}

// ---------------------------------------------------------------------------
// Row L2 norm over 64-element head rows: x f16 [rows][64] -> out f32 [rows]
// ---------------------------------------------------------------------------
__global__ __launch_bounds__(256) void row_norm64(const h16* __restrict__ x,
                                                  float* __restrict__ out) {
  int r = blockIdx.x * 256 + threadIdx.x;
  const h16* p = x + (size_t)r * DKH;
  float s = 0.f;
#pragma unroll
  for (int i = 0; i < DKH; ++i) {
    float v = (float)p[i];
    s += v * v;
  }
  out[r] = sqrtf(s);
}

// ---------------------------------------------------------------------------
// GEMM: Y[4096][1024] = A(f32)[4096][1024] @ WT(f16)[1024][1024]^T + bias
// Block tile 128x64, BK=32, 8 waves as 4x2; each wave 32x32 -> 4 WMMA / K-step.
// MODE 0: f16 head-major qh/kh[(b*16+h)*2048 + t][64]
// MODE 1: f16 transposed vt[(b*16+h)][64][2048]
// MODE 2: f32 flat df[row][1024]
// MODE as template parameter -> branch-free straight-line epilogues.
// ---------------------------------------------------------------------------
template <int MODE>
__global__ __launch_bounds__(256) void gemm_wmma(const float* __restrict__ A,
                                                 const h16*  __restrict__ WT,
                                                 const float* __restrict__ bias,
                                                 h16*  __restrict__ dh,
                                                 float* __restrict__ df) {
  __shared__ h16 As[128 * 32];
  __shared__ h16 Bs[64 * 32];

  int tid = threadIdx.x;
  int w   = tid >> 5;
  int bm  = blockIdx.x * 128;  // 32 M-tiles
  int bn  = blockIdx.y * 64;   // 16 N-tiles
  int wm  = (w & 3) * 32;
  int wn  = (w >> 2) * 32;

  v8f acc00 = {}, acc01 = {}, acc10 = {}, acc11 = {};

  for (int k0 = 0; k0 < DMODEL; k0 += 32) {
    __syncthreads();
    // stage A tile (f32 -> f16): 128x32, 16 elems/thread
    {
      int idx = tid * 16;
      int r = idx >> 5, c = idx & 31;
      const float4* p = (const float4*)(A + (size_t)(bm + r) * DMODEL + k0 + c);
      float4 x0 = p[0], x1 = p[1], x2 = p[2], x3 = p[3];
      h16* d = As + r * 32 + c;
      d[0]  = (h16)x0.x; d[1]  = (h16)x0.y; d[2]  = (h16)x0.z; d[3]  = (h16)x0.w;
      d[4]  = (h16)x1.x; d[5]  = (h16)x1.y; d[6]  = (h16)x1.z; d[7]  = (h16)x1.w;
      d[8]  = (h16)x2.x; d[9]  = (h16)x2.y; d[10] = (h16)x2.z; d[11] = (h16)x2.w;
      d[12] = (h16)x3.x; d[13] = (h16)x3.y; d[14] = (h16)x3.z; d[15] = (h16)x3.w;
    }
    // stage B tile (f16, row-major [n][k]): 64x32, 8 elems/thread
    {
      int idx = tid * 8;
      int r = idx >> 5, c = idx & 31;
      *(uint4*)(Bs + r * 32 + c) =
          *(const uint4*)(WT + (size_t)(bn + r) * DMODEL + k0 + c);
    }
    __syncthreads();

    v16h a0 = frag_a(As + wm * 32, 32);
    v16h a1 = frag_a(As + (wm + 16) * 32, 32);
    v16h b0 = frag_b(Bs + wn * 32, 32);
    v16h b1 = frag_b(Bs + (wn + 16) * 32, 32);
    acc00 = WMMA_F16(a0, b0, acc00);
    acc01 = WMMA_F16(a0, b1, acc01);
    acc10 = WMMA_F16(a1, b0, acc10);
    acc11 = WMMA_F16(a1, b1, acc11);
  }

  int lane = tid & 31;
  int hi   = lane >> 4;
  int j    = lane & 15;
#pragma unroll
  for (int a = 0; a < 2; ++a) {
#pragma unroll
    for (int t = 0; t < 2; ++t) {
      v8f cc = a ? (t ? acc11 : acc10) : (t ? acc01 : acc00);
      int col = bn + wn + t * 16 + j;
      float bv = bias[col];
#pragma unroll
      for (int r = 0; r < 8; ++r) {
        int row = bm + wm + a * 16 + r + 8 * hi;
        float val = cc[r] + bv;
        if (MODE == 2) {
          df[(size_t)row * DMODEL + col] = val;
        } else {
          int b = row >> 11, tt = row & 2047;
          int h = col >> 6,  d  = col & 63;
          int bh = b * NHEADS + h;
          if (MODE == 0)
            dh[((size_t)bh * TSEQ + tt) * DKH + d] = (h16)val;
          else
            dh[(size_t)bh * DKH * TSEQ + (size_t)d * TSEQ + tt] = (h16)val;
        }
      }
    }
  }
}

// ---------------------------------------------------------------------------
// Attention: per (b,h), flash-style over T with online softmax.
// K/V tiles staged by the Tensor Data Mover (double-buffered): wave 0 issues
// the DMA for tile i+1 while all 8 waves compute on tile i, then
// s_wait_tensorcnt 0 + workgroup barrier.
// ---------------------------------------------------------------------------
__global__ __launch_bounds__(256) void attn_wmma(const h16* __restrict__ qh,
                                                 const h16* __restrict__ kh,
                                                 const h16* __restrict__ vt,
                                                 const float* __restrict__ nqg,
                                                 const float* __restrict__ nkg,
                                                 const float* __restrict__ alpha,
                                                 float* __restrict__ outf) {
  __shared__ h16 Qs[128 * 64];        // 16 KB
  __shared__ h16 Ks[2][32 * 64];      //  8 KB (double-buffered keys)
  __shared__ h16 Vts[2][64 * 32];     //  8 KB (double-buffered V^T)
  __shared__ h16 Ps[8 * 16 * 32];     //  8 KB per-wave P scratch
  __shared__ float kn[2][32];

  int tid  = threadIdx.x;
  int w    = tid >> 5;
  int lane = tid & 31;
  int hi   = lane >> 4;
  int j    = lane & 15;

  int bh = blockIdx.x >> 4;
  int q0 = (blockIdx.x & 15) * 128;
  const h16* Q = qh + (size_t)bh * TSEQ * DKH;
  const h16* K = kh + (size_t)bh * TSEQ * DKH;
  const h16* V = vt + (size_t)bh * DKH * TSEQ;

  // Prologue: TDM-load Q block [128][64] and first K/Vt tiles
  if (w == 0) {
    tdm_load_2d_f16(lds_addr_of(Qs),     Q + (size_t)q0 * DKH,
                    DKH, TSEQ, DKH, DKH, 128);
    tdm_load_2d_f16(lds_addr_of(Ks[0]),  K, DKH, TSEQ, DKH, DKH, 32);
    tdm_load_2d_f16(lds_addr_of(Vts[0]), V, TSEQ, DKH, TSEQ, 32, DKH);
    __builtin_amdgcn_s_wait_tensorcnt(0);
  }
  if (tid < 32) kn[0][tid] = nkg[(size_t)bh * TSEQ + tid];
  __syncthreads();

  int wq = w * 16;
  v16h Qf0 = frag_a(Qs + wq * 64, 64);        // d_k  0..31
  v16h Qf1 = frag_a(Qs + wq * 64 + 32, 64);   // d_k 32..63

  float nq[8];
#pragma unroll
  for (int r = 0; r < 8; ++r)
    nq[r] = nqg[(size_t)bh * TSEQ + q0 + wq + r + 8 * hi];

  float sc = 4.0f * alpha[bh & (NHEADS - 1)];

  float m[8], l[8];
  v8f O[4] = {};
#pragma unroll
  for (int r = 0; r < 8; ++r) { m[r] = -1.0e30f; l[r] = 0.f; }

  int cur = 0;
  for (int kb = 0; kb < TSEQ; kb += 32) {
    int nxt = cur ^ 1;
    // Kick off async DMA of the next tile while computing on the current one
    if (kb + 32 < TSEQ) {
      if (w == 0) {
        tdm_load_2d_f16(lds_addr_of(Ks[nxt]),  K + (size_t)(kb + 32) * DKH,
                        DKH, TSEQ, DKH, DKH, 32);
        tdm_load_2d_f16(lds_addr_of(Vts[nxt]), V + kb + 32,
                        TSEQ, DKH, TSEQ, 32, DKH);
      }
      if (tid < 32) kn[nxt][tid] = nkg[(size_t)bh * TSEQ + kb + 32 + tid];
    }

    const h16* Kc = Ks[cur];
    const h16* Vc = Vts[cur];

    // S = Q @ K^T  (two 16-key tiles)
    v8f S0 = {}, S1 = {};
    S0 = WMMA_F16(Qf0, frag_b(Kc, 64), S0);
    S0 = WMMA_F16(Qf1, frag_b(Kc + 32, 64), S0);
    S1 = WMMA_F16(Qf0, frag_b(Kc + 16 * 64, 64), S1);
    S1 = WMMA_F16(Qf1, frag_b(Kc + 16 * 64 + 32, 64), S1);

    float nk0 = kn[cur][j], nk1 = kn[cur][16 + j];
    float p0[8], p1[8];
#pragma unroll
    for (int r = 0; r < 8; ++r) {
      float d0 = nq[r] + nk0; d0 *= d0;
      float d1 = nq[r] + nk1; d1 *= d1;
      float s0 = S0[r] * sc / d0;
      float s1 = S1[r] * sc / d1;
      float mx = fmaxf(s0, s1);
#pragma unroll
      for (int off = 1; off < 16; off <<= 1)
        mx = fmaxf(mx, __shfl_xor(mx, off, 32));
      float mnew = fmaxf(m[r], mx);
      float corr = __expf(m[r] - mnew);
      m[r] = mnew;
      float e0 = __expf(s0 - mnew);
      float e1 = __expf(s1 - mnew);
      float rs = e0 + e1;
#pragma unroll
      for (int off = 1; off < 16; off <<= 1)
        rs += __shfl_xor(rs, off, 32);
      l[r] = l[r] * corr + rs;
#pragma unroll
      for (int g = 0; g < 4; ++g) O[g][r] *= corr;
      p0[r] = e0;
      p1[r] = e1;
    }

    // C-layout P -> A-layout via per-wave LDS scratch
    h16* ps = Ps + w * 512;
#pragma unroll
    for (int r = 0; r < 8; ++r) {
      ps[(r + 8 * hi) * 32 + j]      = (h16)p0[r];
      ps[(r + 8 * hi) * 32 + 16 + j] = (h16)p1[r];
    }
    asm volatile("s_wait_dscnt 0" ::: "memory");  // same-wave LDS RAW
    v16h Pf = frag_a(ps, 32);

    // O += P @ V (V^T tile, 4 column groups of 16)
#pragma unroll
    for (int g = 0; g < 4; ++g)
      O[g] = WMMA_F16(Pf, frag_b(Vc + g * 16 * 32, 32), O[g]);

    if (w == 0) __builtin_amdgcn_s_wait_tensorcnt(0);  // next tiles landed
    __syncthreads();
    cur = nxt;
  }

  // write O / l  -> f32 [B][T][D_MODEL] (concat heads)
  int b = bh >> 4;
  int h = bh & 15;
#pragma unroll
  for (int g = 0; g < 4; ++g) {
#pragma unroll
    for (int r = 0; r < 8; ++r) {
      int t = q0 + wq + r + 8 * hi;
      int d = g * 16 + j;
      outf[((size_t)b * TSEQ + t) * DMODEL + h * DKH + d] = O[g][r] / l[r];
    }
  }
}

// ---------------------------------------------------------------------------
extern "C" void kernel_launch(void* const* d_in, const int* in_sizes, int n_in,
                              void* d_out, int out_size, void* d_ws, size_t ws_size,
                              hipStream_t stream) {
  const float* q     = (const float*)d_in[0];
  const float* k     = (const float*)d_in[1];
  const float* v     = (const float*)d_in[2];
  const float* Wq    = (const float*)d_in[3];
  const float* bq    = (const float*)d_in[4];
  const float* Wk    = (const float*)d_in[5];
  const float* bk    = (const float*)d_in[6];
  const float* Wv    = (const float*)d_in[7];
  const float* bv    = (const float*)d_in[8];
  const float* Wo    = (const float*)d_in[9];
  const float* bo    = (const float*)d_in[10];
  const float* alpha = (const float*)d_in[11];
  float* outp = (float*)d_out;

  char* ws = (char*)d_ws;
  size_t off = 0;
  auto take = [&](size_t bytes) {
    void* p = ws + off;
    off += (bytes + 255) & ~(size_t)255;
    return p;
  };

  h16* WqT = (h16*)take((size_t)DMODEL * DMODEL * 2);
  h16* WkT = (h16*)take((size_t)DMODEL * DMODEL * 2);
  h16* WvT = (h16*)take((size_t)DMODEL * DMODEL * 2);
  h16* WoT = (h16*)take((size_t)DMODEL * DMODEL * 2);
  h16* qhB = (h16*)take((size_t)BHTOT * TSEQ * DKH * 2);
  h16* khB = (h16*)take((size_t)BHTOT * TSEQ * DKH * 2);
  h16* vtB = (h16*)take((size_t)BHTOT * DKH * TSEQ * 2);
  float* nqB = (float*)take((size_t)BHTOT * TSEQ * 4);
  float* nkB = (float*)take((size_t)BHTOT * TSEQ * 4);
  float* atB = (float*)take((size_t)MROWS * DMODEL * 4);

  // 1) weight transpose + f16 convert
  wconvT<<<DMODEL * DMODEL / 256, 256, 0, stream>>>(Wq, WqT);
  wconvT<<<DMODEL * DMODEL / 256, 256, 0, stream>>>(Wk, WkT);
  wconvT<<<DMODEL * DMODEL / 256, 256, 0, stream>>>(Wv, WvT);
  wconvT<<<DMODEL * DMODEL / 256, 256, 0, stream>>>(Wo, WoT);

  // 2) QKV projections (WMMA GEMM)
  dim3 ggrid(MROWS / 128, DMODEL / 64);
  gemm_wmma<0><<<ggrid, 256, 0, stream>>>(q, WqT, bq, qhB, nullptr);
  gemm_wmma<0><<<ggrid, 256, 0, stream>>>(k, WkT, bk, khB, nullptr);
  gemm_wmma<1><<<ggrid, 256, 0, stream>>>(v, WvT, bv, vtB, nullptr);

  // 3) per-row head norms
  row_norm64<<<BHTOT * TSEQ / 256, 256, 0, stream>>>(qhB, nqB);
  row_norm64<<<BHTOT * TSEQ / 256, 256, 0, stream>>>(khB, nkB);

  // 4) attention (flash-style, WMMA + TDM double buffering)
  attn_wmma<<<BHTOT * (TSEQ / 128), 256, 0, stream>>>(qhB, khB, vtB, nqB, nkB,
                                                      alpha, atB);

  // 5) output projection -> d_out (f32)
  gemm_wmma<2><<<ggrid, 256, 0, stream>>>(atB, WoT, bo, nullptr, outp);
}
